// HeterRelGraphConv_21809843929179
// MI455X (gfx1250) — matrix operand order repaired and checked by reading, not compiled, not run on hardware
//
#include <hip/hip_runtime.h>
#include <hip/hip_bf16.h>

// R-GCN basis-decomposition layer for MI455X (gfx1250, wave32, WMMA).
//
// Never materialize xw[16,100000,64] (410MB like the reference). Counting-sort
// edges by relation; each wave then processes 8 relation-uniform 16-edge tiles
// (16x64 @ 64x64 bf16 WMMA each, 8 x v_wmma_f32_16x16x32_bf16 per tile) while
// holding the relation's B fragments in registers. x lives as an L2-resident
// bf16 copy (12.8MB << 192MB L2); messages scatter-add with
// global_atomic_add_f32 into the L2-resident output.

#define N_NODES  100000
#define N_EDGES  1600000
#define FEAT     64
#define NUM_RELS 16
#define NUM_BASES 8
#define TILES_PER_WAVE 8          // 8 x 16 = 128 edges per wave, B frags stay in VGPRs

typedef __attribute__((ext_vector_type(16))) __bf16 v16bf;
typedef __attribute__((ext_vector_type(8)))  __bf16 v8bf;
typedef __attribute__((ext_vector_type(8)))  float  v8f;

// ---- workspace layout (bytes) ----
static constexpr size_t WS_XBF  = 0;                       // N_NODES*64 u16 = 12,800,000
static constexpr size_t WS_WSWZ = 12800000;                // 17*64*64 u16   =    139,264
static constexpr size_t WS_CNT  = 12939264;                // 16 int
static constexpr size_t WS_OFF  = WS_CNT  + 64;            // 17 int
static constexpr size_t WS_TOFF = WS_OFF  + 68;            // 17 int (super-tile offsets)
static constexpr size_t WS_CUR  = WS_TOFF + 68;            // 16 int
static constexpr size_t WS_PERM = 12939536;                // N_EDGES int = 6,400,000
// total ~19.34 MB

__device__ __forceinline__ unsigned short f32_to_bf16(float f) {
    unsigned u = __float_as_uint(f);
    u += 0x7FFFu + ((u >> 16) & 1u);   // round-to-nearest-even
    return (unsigned short)(u >> 16);
}

// A-fragment for v_wmma_f32_16x16x32_bf16 (16-bit A 16x32), contiguous k order used here, matching B (not the ISA table order):
// lanes 0-15: row=lane,  elems 0..15 = K 0..15
// lanes 16-31: row=lane-16, elems 0..15 = K 16..31
__device__ __forceinline__ v16bf load_a_frag(const unsigned short* rowp, int kb, int half) {
    const v8bf lo = *(const v8bf*)(rowp + kb * 32 + half * 16);
    const v8bf hi = *(const v8bf*)(rowp + kb * 32 + half * 16 + 8);
    return __builtin_shufflevector(lo, hi, 0,1,2,3,4,5,6,7,8,9,10,11,12,13,14,15);
}

// ---------------- stage 0: x -> bf16, zero histogram ----------------
__global__ void k_convert_x(const float* __restrict__ x,
                            unsigned short* __restrict__ xbf,
                            int* __restrict__ cnt) {
    int i = blockIdx.x * blockDim.x + threadIdx.x;
    if (i < N_NODES * FEAT) xbf[i] = f32_to_bf16(x[i]);
    if (blockIdx.x == 0 && threadIdx.x < NUM_RELS) cnt[threadIdx.x] = 0;
}

// ---------------- stage 1: compose W[r]=sum_b comp*basis, swizzle to B-frag layout
// B-fragment layout (32x16 bf16 B): lanes 0-15: col=lane, elems = K 0..15;
// lanes 16-31: col=lane-16, elems = K 16..31. Stored as [m][kb][nt][lane][elem].
__global__ void k_compose(const float* __restrict__ weight,
                          const float* __restrict__ w_comp,
                          const float* __restrict__ loop_w,
                          unsigned short* __restrict__ wswz) {
    int m = blockIdx.x;  // 0..15 relations, 16 = loop_weight
    for (int e = threadIdx.x; e < FEAT * FEAT; e += blockDim.x) {
        int i = e >> 6, o = e & 63;   // i = K (in feat), o = N (out feat)
        float v;
        if (m < NUM_RELS) {
            v = 0.f;
            #pragma unroll
            for (int b = 0; b < NUM_BASES; ++b)
                v += w_comp[m * NUM_BASES + b] * weight[(b * FEAT + i) * FEAT + o];
        } else {
            v = loop_w[e];
        }
        int kb   = i >> 5, kr = i & 31;
        int nt   = o >> 4, col = o & 15;
        int lane = (kr < 16) ? col : (16 + col);
        int elem = kr & 15;
        wswz[(((size_t)(m * 8 + kb * 4 + nt) * 32 + lane) * 16) + elem] = f32_to_bf16(v);
    }
}

// ---------------- stage 2-4: counting sort of edges by relation ----------------
__global__ void k_hist(const int* __restrict__ etype, int* __restrict__ cnt) {
    int e = blockIdx.x * blockDim.x + threadIdx.x;
    if (e < N_EDGES) atomicAdd(&cnt[etype[e]], 1);
}

__global__ void k_scan(const int* __restrict__ cnt, int* __restrict__ off,
                       int* __restrict__ toff, int* __restrict__ cur) {
    if (threadIdx.x == 0 && blockIdx.x == 0) {
        const int SUPER = 16 * TILES_PER_WAVE;   // edges per wave
        int o = 0, t = 0;
        for (int r = 0; r < NUM_RELS; ++r) {
            off[r] = o; toff[r] = t; cur[r] = o;
            o += cnt[r];
            t += (cnt[r] + SUPER - 1) / SUPER;
        }
        off[NUM_RELS] = o; toff[NUM_RELS] = t;
    }
}

__global__ void k_scatter(const int* __restrict__ etype, int* __restrict__ cur,
                          int* __restrict__ perm) {
    int e = blockIdx.x * blockDim.x + threadIdx.x;
    if (e < N_EDGES) {
        int p = atomicAdd(&cur[etype[e]], 1);
        perm[p] = e;
    }
}

// ---------------- stage 5: self-loop GEMM h0 = x @ loop_w + bias ----------------
__global__ void k_selfloop(const unsigned short* __restrict__ xbf,
                           const unsigned short* __restrict__ wswz,
                           const float* __restrict__ bias,
                           float* __restrict__ out) {
    int wave = (blockIdx.x * blockDim.x + threadIdx.x) >> 5;
    int lane = threadIdx.x & 31;
    if (wave >= N_NODES / 16) return;           // wave-uniform exit (EXEC stays full for WMMA)
    int half = lane >> 4, col = lane & 15;
    int node0 = wave * 16;
    const unsigned short* rowp = xbf + (size_t)(node0 + col) * FEAT;  // A row = lane&15
    const v16bf* B = (const v16bf*)wswz + (size_t)NUM_RELS * 8 * 32;  // m==16 slot
    v8f acc[4] = {};
    #pragma unroll
    for (int kb = 0; kb < 2; ++kb) {
        v16bf a = load_a_frag(rowp, kb, half);
        #pragma unroll
        for (int nt = 0; nt < 4; ++nt) {
            v16bf b = B[(kb * 4 + nt) * 32 + lane];
            acc[nt] = __builtin_amdgcn_wmma_f32_16x16x32_bf16(
                false, a, false, b, (short)0, acc[nt], false, false);
        }
    }
    #pragma unroll
    for (int nt = 0; nt < 4; ++nt) {
        float bv = bias[nt * 16 + col];
        #pragma unroll
        for (int v = 0; v < 8; ++v) {
            int row = v + half * 8;             // C/D layout: vgpr v -> row, lane half -> +8
            out[(size_t)(node0 + row) * FEAT + nt * 16 + col] = acc[nt][v] + bv;
        }
    }
}

// ---------------- stage 6: edge messages ----------------
// Each wave owns one relation-uniform super-tile of up to 128 edges, keeps the
// relation's 8 B-fragments (8KB) in VGPRs, and loops over 8 sub-tiles of 16
// edges: gather A (two b128 per lane from L2-resident bf16 x), 8 WMMAs,
// scatter rows with f32 atomics.
__global__ void k_edges(const unsigned short* __restrict__ xbf,
                        const unsigned short* __restrict__ wswz,
                        const int* __restrict__ esrc,
                        const int* __restrict__ edst,
                        const int* __restrict__ perm,
                        const int* __restrict__ off,
                        const int* __restrict__ toff,
                        float* __restrict__ out) {
    int sb   = (blockIdx.x * blockDim.x + threadIdx.x) >> 5;
    int lane = threadIdx.x & 31;
    if (sb >= toff[NUM_RELS]) return;            // wave-uniform
    // locate relation: toff[r] <= sb < toff[r+1]
    int r = 0;
    while (sb >= toff[r + 1]) ++r;
    int base = off[r] + (sb - toff[r]) * 16 * TILES_PER_WAVE;
    int rend = off[r + 1];

    int half = lane >> 4, col = lane & 15;

    // preload the relation's B fragments into registers (reused for all tiles)
    const v16bf* B = (const v16bf*)wswz + (size_t)r * 8 * 32;
    v16bf bfrag[8];
    #pragma unroll
    for (int f = 0; f < 8; ++f) bfrag[f] = B[f * 32 + lane];

    for (int t = 0; t < TILES_PER_WAVE; ++t) {
        int ebase = base + t * 16;
        if (ebase >= rend) break;                // wave-uniform
        int ecount = min(16, rend - ebase);

        int m  = col;                            // A row handled by this lane
        int me = (m < ecount) ? m : 0;           // clamp padded rows to a valid edge
        int e  = perm[ebase + me];
        int s  = esrc[e];
        int d  = edst[e];

        const unsigned short* rowp = xbf + (size_t)s * FEAT;   // random 16B loads, L2-resident

        v8f acc[4] = {};
        #pragma unroll
        for (int kb = 0; kb < 2; ++kb) {
            v16bf a = load_a_frag(rowp, kb, half);
            #pragma unroll
            for (int nt = 0; nt < 4; ++nt) {
                acc[nt] = __builtin_amdgcn_wmma_f32_16x16x32_bf16(
                    false, a, false, bfrag[kb * 4 + nt], (short)0, acc[nt], false, false);
            }
        }
        // scatter-add rows to destination nodes (output is 25.6MB -> L2 atomics)
        #pragma unroll
        for (int nt = 0; nt < 4; ++nt) {
            #pragma unroll
            for (int v = 0; v < 8; ++v) {
                int row  = v + half * 8;
                int drow = __shfl(d, row, 32);   // dst of the edge owning this C row
                if (row < ecount)
                    atomicAdd(&out[(size_t)drow * FEAT + nt * 16 + col], acc[nt][v]);
            }
        }
    }
}

// ---------------- stage 7: ReLU ----------------
__global__ void k_relu(float* __restrict__ out) {
    int i = blockIdx.x * blockDim.x + threadIdx.x;
    if (i < N_NODES * FEAT) out[i] = fmaxf(out[i], 0.f);
}

extern "C" void kernel_launch(void* const* d_in, const int* in_sizes, int n_in,
                              void* d_out, int out_size, void* d_ws, size_t ws_size,
                              hipStream_t stream) {
    const float* x      = (const float*)d_in[0];
    const int*   esrc   = (const int*)  d_in[1];
    const int*   edst   = (const int*)  d_in[2];
    const int*   etype  = (const int*)  d_in[3];
    const float* weight = (const float*)d_in[4];
    const float* w_comp = (const float*)d_in[5];
    const float* h_bias = (const float*)d_in[6];
    const float* loop_w = (const float*)d_in[7];
    float* out = (float*)d_out;

    char* ws = (char*)d_ws;
    unsigned short* xbf  = (unsigned short*)(ws + WS_XBF);
    unsigned short* wswz = (unsigned short*)(ws + WS_WSWZ);
    int* cnt  = (int*)(ws + WS_CNT);
    int* off  = (int*)(ws + WS_OFF);
    int* toff = (int*)(ws + WS_TOFF);
    int* cur  = (int*)(ws + WS_CUR);
    int* perm = (int*)(ws + WS_PERM);

    k_convert_x<<<(N_NODES * FEAT + 255) / 256, 256, 0, stream>>>(x, xbf, cnt);
    k_compose  <<<NUM_RELS + 1, 256, 0, stream>>>(weight, w_comp, loop_w, wswz);
    k_hist     <<<(N_EDGES + 255) / 256, 256, 0, stream>>>(etype, cnt);
    k_scan     <<<1, 32, 0, stream>>>(cnt, off, toff, cur);
    k_scatter  <<<(N_EDGES + 255) / 256, 256, 0, stream>>>(etype, cur, perm);

    // self-loop initializes d_out (stores), then edges atomically accumulate.
    int sl_waves = N_NODES / 16;                             // 6250 tiles
    k_selfloop <<<(sl_waves * 32 + 255) / 256, 256, 0, stream>>>(xbf, wswz, h_bias, out);
    int max_super = N_EDGES / (16 * TILES_PER_WAVE) + NUM_RELS;  // >= sum_r ceil(cnt_r/128)
    k_edges    <<<(max_super * 32 + 255) / 256, 256, 0, stream>>>(
        xbf, wswz, esrc, edst, perm, off, toff, out);
    k_relu     <<<(N_NODES * FEAT + 255) / 256, 256, 0, stream>>>(out);
}
